// GenuineTransformer_16432544874500
// MI455X (gfx1250) — compile-verified, matrix operand order checked by reading, not run on hardware
//
#include <hip/hip_runtime.h>

typedef __attribute__((ext_vector_type(16))) __bf16 v16bf;
typedef __attribute__((ext_vector_type(8)))  float  v8f;
typedef __attribute__((ext_vector_type(4)))  unsigned int u32x4;
typedef __attribute__((ext_vector_type(8)))  int i32x8;
typedef __attribute__((ext_vector_type(4)))  int i32x4;
typedef unsigned short u16;
typedef unsigned int   u32;

#define D_MODEL   1024
#define N_HEADS   16
#define HEAD_DIM  64
#define BATCH     16
#define SEQ       128
#define TOKENS    (BATCH*SEQ)     // 2048
#define D_FF      4096
#define VOCAB     32000
#define LOG2E_F   1.44269504f

// ---------------------------------------------------------------- helpers
__device__ __forceinline__ u16 f2bf(float f) {
  u32 u = __float_as_uint(f);
  u += 0x7FFFu + ((u >> 16) & 1u);       // round-to-nearest-even
  return (u16)(u >> 16);
}

// Load a 16x32 bf16 WMMA fragment (A layout; B uses [n][k] so same loader).
// ISA 7.12.2: lanes 0-15 row M, k 0-7 (v0-3) and 16-23 (v4-7); lanes 16-31 k+8.
__device__ __forceinline__ v16bf ld_frag(const u16* base, int stride, int row0,
                                         int k0, int lane) {
  const u16* p = base + (row0 + (lane & 15)) * stride + k0 + ((lane >> 4) << 3);
  union { uint4 q[2]; v16bf v; } u;
  u.q[0] = *(const uint4*)(p);        // k = k0 + half*8 .. +7
  u.q[1] = *(const uint4*)(p + 16);   // k = k0 + 16 + half*8 .. +7
  return u.v;
}

__device__ __forceinline__ v8f wmma_bf16(v16bf a, v16bf b, v8f c) {
  return __builtin_amdgcn_wmma_f32_16x16x32_bf16(false, a, false, b, (short)0,
                                                 c, false, false);
}

// Tensor Data Mover: 2D tile (64 cols x 128 rows of bf16) global -> LDS,
// with LDS padding of 4 DWORDs every 32 DWORDs (row stride 128B -> 144B = 72 elems).
// D# per ISA cdna5 ch.8: group0 {count=1 | lds_addr | global_addr | type=2},
// group1 {data_size=1(2B), pad_enable, pad_interval=4(32dw), pad_amount=3(4dw),
//         tensor_dim0=K, tensor_dim1=rows, tile_dim0=64, tile_dim1=128, stride0=K}.
// This toolchain exposes the 6-arg builtin (extra i32x8 group, zero-filled).
__device__ __forceinline__ void tdm_load_tile(unsigned lds_byte_off,
                                              const u16* gptr, int K, int rows) {
  unsigned long long ga = (unsigned long long)(uintptr_t)gptr;
  u32x4 g0 = { 1u,
               lds_byte_off,
               (u32)(ga & 0xFFFFFFFFull),
               ((u32)((ga >> 32) & 0x1FFFFFFull)) | 0x80000000u };
  u32 td0 = (u32)K, td1 = (u32)rows;
  i32x8 g1 = { (int)((1u << 16) | (1u << 20) | (4u << 22) | (3u << 25)),
               (int)((td0 & 0xFFFFu) << 16),
               (int)((td0 >> 16) | ((td1 & 0xFFFFu) << 16)),
               (int)((td1 >> 16) | (64u << 16)),
               (int)128,
               (int)(u32)K,
               0, 0 };
  i32x4 gz4 = { 0, 0, 0, 0 };
  i32x8 gz8 = { 0, 0, 0, 0, 0, 0, 0, 0 };
  __builtin_amdgcn_tensor_load_to_lds(g0, g1, gz4, gz4, gz8, 0);
}

// ---------------------------------------------------------------- embed
__global__ __launch_bounds__(256) void embed_kernel(const int* __restrict__ x,
                                                    const float* __restrict__ emb,
                                                    float* __restrict__ h) {
  int row = blockIdx.x;
  int tok = x[row];
  int c = threadIdx.x * 4;
  float4 v = *(const float4*)&emb[(size_t)tok * D_MODEL + c];
  *(float4*)&h[(size_t)row * D_MODEL + c] = v;
}

// ---------------------------------------------------------------- layernorm -> bf16
__global__ __launch_bounds__(256) void ln_kernel(const float* __restrict__ x,
                                                 const float* __restrict__ w,
                                                 const float* __restrict__ b,
                                                 u16* __restrict__ out) {
  __shared__ float red[256];
  int row = blockIdx.x, t = threadIdx.x;
  const float* xr = x + (size_t)row * D_MODEL;
  float4 v = *(const float4*)&xr[t * 4];
  float s = v.x + v.y + v.z + v.w;
  red[t] = s; __syncthreads();
  for (int o = 128; o > 0; o >>= 1) { if (t < o) red[t] += red[t + o]; __syncthreads(); }
  float mean = red[0] * (1.0f / D_MODEL);
  __syncthreads();
  float dx = v.x - mean, dy = v.y - mean, dz = v.z - mean, dw = v.w - mean;
  red[t] = dx*dx + dy*dy + dz*dz + dw*dw; __syncthreads();
  for (int o = 128; o > 0; o >>= 1) { if (t < o) red[t] += red[t + o]; __syncthreads(); }
  float rs = rsqrtf(red[0] * (1.0f / D_MODEL) + 1e-5f);
  float4 wv = *(const float4*)&w[t * 4];
  float4 bv = *(const float4*)&b[t * 4];
  uint2 pk;
  pk.x = (u32)f2bf(dx * rs * wv.x + bv.x) | ((u32)f2bf(dy * rs * wv.y + bv.y) << 16);
  pk.y = (u32)f2bf(dz * rs * wv.z + bv.z) | ((u32)f2bf(dw * rs * wv.w + bv.w) << 16);
  *(uint2*)&out[(size_t)row * D_MODEL + t * 4] = pk;
}

// ---------------------------------------------------------------- f32 -> bf16 cast
__global__ __launch_bounds__(256) void cast_kernel(const float* __restrict__ in,
                                                   u16* __restrict__ out) {
  int i = (blockIdx.x * 256 + threadIdx.x) * 4;
  float4 v = *(const float4*)&in[i];
  uint2 pk;
  pk.x = (u32)f2bf(v.x) | ((u32)f2bf(v.y) << 16);
  pk.y = (u32)f2bf(v.z) | ((u32)f2bf(v.w) << 16);
  *(uint2*)&out[i] = pk;
}

// ---------------------------------------------------------------- RoPE + relayout to (B,H,S,D) bf16
__global__ __launch_bounds__(256) void rope_kernel(const float* __restrict__ qf,
                                                   const float* __restrict__ kf,
                                                   const float* __restrict__ vf,
                                                   u16* __restrict__ qb,
                                                   u16* __restrict__ kb,
                                                   u16* __restrict__ vb) {
  int idx = blockIdx.x * 256 + threadIdx.x;     // 0 .. B*S*H*32-1
  int i = idx & 31;
  int h = (idx >> 5) & 15;
  int s = (idx >> 9) & 127;
  int b = idx >> 16;
  float f = (float)s * __expf(-(float)(2 * i) * (9.210340371976184f / 64.0f));
  float c, sn;
  __sincosf(f, &sn, &c);
  size_t gi = ((size_t)(b * SEQ + s)) * D_MODEL + h * HEAD_DIM + 2 * i;
  size_t go = ((size_t)((b * N_HEADS + h) * SEQ + s)) * HEAD_DIM + 2 * i;
  float qr = qf[gi], qi = qf[gi + 1];
  qb[go]     = f2bf(qr * c - qi * sn);
  qb[go + 1] = f2bf(qr * sn + qi * c);
  float kr = kf[gi], ki = kf[gi + 1];
  kb[go]     = f2bf(kr * c - ki * sn);
  kb[go + 1] = f2bf(kr * sn + ki * c);
  vb[go]     = f2bf(vf[gi]);
  vb[go + 1] = f2bf(vf[gi + 1]);
}

// ================================================================ TDM GEMM
// C[M,N] = A[M,K](bf16) @ Wb[N,K](bf16)^T, 128x128 block, 8 waves (64x32 each),
// K-step 64, double-buffered tensor_load_to_lds staging (TENSORcnt pipelined).
// LDS elem layout: [A0 | B0 | A1 | B1], each TTILE elems.
#define TTILE (128 * 72)                 // elems per LDS tile (padded stride 72)
#define TDM_SMEM_BYTES ((size_t)4 * TTILE * 2)   // 73728 B

template<bool BIAS, bool GELU, bool RES, bool OUTBF>
__global__ __launch_bounds__(256) void gemm_tdm_kernel(
    const u16* __restrict__ A, const u16* __restrict__ Wb,
    const float* __restrict__ bias, const float* __restrict__ res,
    void* __restrict__ outp, int M, int N, int K) {
  extern __shared__ u16 sm[];
  unsigned ldsBase = (unsigned)(uintptr_t)sm;    // LDS byte offset of dynamic base

  int t = threadIdx.x, lane = t & 31, w = t >> 5;
  int m0 = blockIdx.x * 128, n0 = blockIdx.y * 128;
  int wm = (w >> 2) * 64, wn = (w & 3) * 32;
  int nst = K >> 6;

  if (w == 0) {
    tdm_load_tile(ldsBase,                 A  + (size_t)m0 * K, K, M);
    tdm_load_tile(ldsBase + 2u * TTILE,    Wb + (size_t)n0 * K, K, N);  // B0 at TTILE elems = 2*TTILE bytes
  }

  v8f acc[4][2];
#pragma unroll
  for (int i = 0; i < 4; ++i)
#pragma unroll
    for (int c = 0; c < 2; ++c)
#pragma unroll
      for (int j = 0; j < 8; ++j) acc[i][c][j] = 0.0f;

  for (int s = 0; s < nst; ++s) {
    int par = s & 1;
    const u16* Ab = sm + par * (2 * TTILE);
    const u16* Bb = sm + TTILE + par * (2 * TTILE);
    if (w == 0) {
      if (s + 1 < nst) {
        unsigned pb = (unsigned)(((s + 1) & 1) * (4 * TTILE));   // byte offset of next pair
        tdm_load_tile(ldsBase + pb,               A  + (size_t)m0 * K + (s + 1) * 64, K, M);
        tdm_load_tile(ldsBase + pb + 2u * TTILE,  Wb + (size_t)n0 * K + (s + 1) * 64, K, N);
        __builtin_amdgcn_s_wait_tensorcnt(2);   // stage-s tiles landed
      } else {
        __builtin_amdgcn_s_wait_tensorcnt(0);
      }
    }
    __syncthreads();
#pragma unroll
    for (int kh = 0; kh < 2; ++kh) {
      int k0 = kh * 32;
      v16bf b0f = ld_frag(Bb, 72, wn,      k0, lane);
      v16bf b1f = ld_frag(Bb, 72, wn + 16, k0, lane);
#pragma unroll
      for (int i = 0; i < 4; ++i) {
        v16bf af = ld_frag(Ab, 72, wm + i * 16, k0, lane);
        acc[i][0] = wmma_bf16(af, b0f, acc[i][0]);
        acc[i][1] = wmma_bf16(af, b1f, acc[i][1]);
      }
    }
    __syncthreads();   // all frag reads done before next DMA overwrites buffer
  }

  int half = lane >> 4, ln = lane & 15;
#pragma unroll
  for (int i = 0; i < 4; ++i)
#pragma unroll
    for (int c = 0; c < 2; ++c)
#pragma unroll
      for (int r = 0; r < 8; ++r) {
        int m = m0 + wm + i * 16 + r + half * 8;
        int n = n0 + wn + c * 16 + ln;
        float v = acc[i][c][r];
        if (BIAS) v += bias[n];
        if (GELU) v = 0.5f * v * (1.0f + erff(v * 0.70710678118654752f));
        if (RES)  v += res[(size_t)m * N + n];
        if (OUTBF) ((u16*)outp)[(size_t)m * N + n] = f2bf(v);
        else       ((float*)outp)[(size_t)m * N + n] = v;
      }
}

// ================================================================ fallback GEMM (f32 weights)
#define BM 64
#define BN 64
#define LDST 40

template<bool BIAS, bool GELU, bool RES, bool OUTBF>
__global__ __launch_bounds__(128) void gemm_bf16_kernel(
    const u16* __restrict__ A, const float* __restrict__ W,
    const float* __restrict__ bias, const float* __restrict__ res,
    void* __restrict__ outp, int M, int N, int K) {
  __shared__ __align__(16) u16 As[BM * LDST];
  __shared__ __align__(16) u16 Bs[BN * LDST];
  int t = threadIdx.x;
  int lane = t & 31, w = t >> 5;
  int m0 = blockIdx.x * BM, n0 = blockIdx.y * BN;
  int wm = (w >> 1) * 32, wn = (w & 1) * 32;

  v8f acc[2][2];
#pragma unroll
  for (int a = 0; a < 2; ++a)
#pragma unroll
    for (int c = 0; c < 2; ++c)
#pragma unroll
      for (int j = 0; j < 8; ++j) acc[a][c][j] = 0.0f;

  for (int k0 = 0; k0 < K; k0 += 32) {
    __syncthreads();
#pragma unroll
    for (int j = 0; j < 2; ++j) {
      int ch = t + j * 128;
      int r = ch >> 2, c8 = (ch & 3) << 3;
      *(uint4*)&As[r * LDST + c8] =
          *(const uint4*)&A[(size_t)(m0 + r) * K + k0 + c8];
    }
#pragma unroll
    for (int j = 0; j < 4; ++j) {
      int ch = t + j * 128;
      int r = ch >> 3, c4 = (ch & 7) << 2;
      float4 f = *(const float4*)&W[(size_t)(n0 + r) * K + k0 + c4];
      uint2 pk;
      pk.x = (u32)f2bf(f.x) | ((u32)f2bf(f.y) << 16);
      pk.y = (u32)f2bf(f.z) | ((u32)f2bf(f.w) << 16);
      *(uint2*)&Bs[r * LDST + c4] = pk;
    }
    __syncthreads();
    v16bf a0 = ld_frag(As, LDST, wm,      0, lane);
    v16bf a1 = ld_frag(As, LDST, wm + 16, 0, lane);
    v16bf b0 = ld_frag(Bs, LDST, wn,      0, lane);
    v16bf b1 = ld_frag(Bs, LDST, wn + 16, 0, lane);
    acc[0][0] = wmma_bf16(a0, b0, acc[0][0]);
    acc[0][1] = wmma_bf16(a0, b1, acc[0][1]);
    acc[1][0] = wmma_bf16(a1, b0, acc[1][0]);
    acc[1][1] = wmma_bf16(a1, b1, acc[1][1]);
  }

  int half = lane >> 4, ln = lane & 15;
#pragma unroll
  for (int tr = 0; tr < 2; ++tr)
#pragma unroll
    for (int tc = 0; tc < 2; ++tc)
#pragma unroll
      for (int r = 0; r < 8; ++r) {
        int m = m0 + wm + tr * 16 + r + half * 8;
        int n = n0 + wn + tc * 16 + ln;
        float v = acc[tr][tc][r];
        if (BIAS) v += bias[n];
        if (GELU) v = 0.5f * v * (1.0f + erff(v * 0.70710678118654752f));
        if (RES)  v += res[(size_t)m * N + n];
        if (OUTBF) ((u16*)outp)[(size_t)m * N + n] = f2bf(v);
        else       ((float*)outp)[(size_t)m * N + n] = v;
      }
}

// ---------------------------------------------------------------- fused attention per (b,h)
#define QS_STRIDE 72
#define VT_STRIDE 136
#define AT_STRIDE 136
#define SM_QS   (SEQ * QS_STRIDE)
#define SM_VT   (HEAD_DIM * VT_STRIDE)
#define SM_AT   (8 * 16 * AT_STRIDE)
#define ATTN_SMEM_BYTES ((size_t)(2 * SM_QS + SM_VT + SM_AT) * 2)   // 89088 B

__global__ __launch_bounds__(256) void attn_kernel(
    const u16* __restrict__ qb, const u16* __restrict__ kb,
    const u16* __restrict__ vb, u16* __restrict__ ob,
    float* __restrict__ ent) {
  extern __shared__ u16 smem[];
  u16* Qs = smem;
  u16* Ks = smem + SM_QS;
  u16* Vt = smem + 2 * SM_QS;
  u16* At = smem + 2 * SM_QS + SM_VT;

  int bh = blockIdx.x;
  int b = bh >> 4, h = bh & 15;
  int t = threadIdx.x, lane = t & 31, w = t >> 5;
  const u16* qg = qb + (size_t)bh * SEQ * HEAD_DIM;
  const u16* kg = kb + (size_t)bh * SEQ * HEAD_DIM;
  const u16* vg = vb + (size_t)bh * SEQ * HEAD_DIM;

#pragma unroll
  for (int j = 0; j < 4; ++j) {
    int ch = t + j * 256;
    int r = ch >> 3, c = (ch & 7) << 3;
    *(uint4*)&Qs[r * QS_STRIDE + c] = *(const uint4*)&qg[r * HEAD_DIM + c];
    *(uint4*)&Ks[r * QS_STRIDE + c] = *(const uint4*)&kg[r * HEAD_DIM + c];
  }
#pragma unroll
  for (int j = 0; j < 32; ++j) {
    int e = t + j * 256;
    int s = e >> 6, d = e & 63;
    Vt[d * VT_STRIDE + s] = vg[e];
  }
  __syncthreads();

  int q0 = w * 16;
  v8f accS[8];
#pragma unroll
  for (int kt = 0; kt < 8; ++kt)
#pragma unroll
    for (int j = 0; j < 8; ++j) accS[kt][j] = 0.0f;

#pragma unroll
  for (int kt = 0; kt < 8; ++kt) {
    accS[kt] = wmma_bf16(ld_frag(Qs, QS_STRIDE, q0, 0, lane),
                         ld_frag(Ks, QS_STRIDE, kt * 16, 0, lane), accS[kt]);
    accS[kt] = wmma_bf16(ld_frag(Qs, QS_STRIDE, q0, 32, lane),
                         ld_frag(Ks, QS_STRIDE, kt * 16, 32, lane), accS[kt]);
  }
#pragma unroll
  for (int kt = 0; kt < 8; ++kt)
#pragma unroll
    for (int j = 0; j < 8; ++j) accS[kt][j] *= 0.125f;

  u16* Aw = At + w * 16 * AT_STRIDE;
  int half = lane >> 4, ln = lane & 15;

#pragma unroll
  for (int r = 0; r < 8; ++r) {
    int m = r + half * 8;
    float mx = accS[0][r];
#pragma unroll
    for (int kt = 1; kt < 8; ++kt) mx = fmaxf(mx, accS[kt][r]);
    for (int msk = 1; msk < 16; msk <<= 1) mx = fmaxf(mx, __shfl_xor(mx, msk, 32));
    float pv[8], Z = 0.0f;
#pragma unroll
    for (int kt = 0; kt < 8; ++kt) {
      float e = accS[kt][r] - mx;
      accS[kt][r] = e;
      pv[kt] = expf(e);
      Z += pv[kt];
    }
    for (int msk = 1; msk < 16; msk <<= 1) Z += __shfl_xor(Z, msk, 32);
    float logZ = logf(Z), rinv = 1.0f / Z;
    float tt = 0.0f;
#pragma unroll
    for (int kt = 0; kt < 8; ++kt) {
      tt += pv[kt] * (logZ - accS[kt][r]);
      Aw[m * AT_STRIDE + kt * 16 + ln] = f2bf(pv[kt] * rinv);
    }
    for (int msk = 1; msk < 16; msk <<= 1) tt += __shfl_xor(tt, msk, 32);
    if (ln == 0) {
      int q = q0 + m;
      ent[((size_t)(b * SEQ + q)) * N_HEADS + h] = tt * rinv * LOG2E_F;
    }
  }
  __syncthreads();

  v8f accO[4];
#pragma unroll
  for (int nt = 0; nt < 4; ++nt)
#pragma unroll
    for (int j = 0; j < 8; ++j) accO[nt][j] = 0.0f;
#pragma unroll
  for (int nt = 0; nt < 4; ++nt)
#pragma unroll
    for (int ks = 0; ks < 4; ++ks)
      accO[nt] = wmma_bf16(ld_frag(Aw, AT_STRIDE, 0, ks * 32, lane),
                           ld_frag(Vt, VT_STRIDE, nt * 16, ks * 32, lane),
                           accO[nt]);
#pragma unroll
  for (int nt = 0; nt < 4; ++nt)
#pragma unroll
    for (int r = 0; r < 8; ++r) {
      int q = q0 + r + half * 8;
      int n = nt * 16 + ln;
      ob[((size_t)(b * SEQ + q)) * D_MODEL + h * HEAD_DIM + n] = f2bf(accO[nt][r]);
    }
}

// ---------------------------------------------------------------- launcher
extern "C" void kernel_launch(void* const* d_in, const int* in_sizes, int n_in,
                              void* d_out, int out_size, void* d_ws, size_t ws_size,
                              hipStream_t stream) {
  (void)in_sizes; (void)n_in; (void)out_size;
  const int*   x    = (const int*)d_in[0];
  const float* emb  = (const float*)d_in[1];
  const float* Wq   = (const float*)d_in[2];
  const float* Wk   = (const float*)d_in[3];
  const float* Wv   = (const float*)d_in[4];
  const float* Wo   = (const float*)d_in[5];
  const float* ln1w = (const float*)d_in[6];
  const float* ln1b = (const float*)d_in[7];
  const float* W1   = (const float*)d_in[8];
  const float* b1   = (const float*)d_in[9];
  const float* W2   = (const float*)d_in[10];
  const float* b2   = (const float*)d_in[11];
  const float* ln2w = (const float*)d_in[12];
  const float* ln2b = (const float*)d_in[13];
  const float* Wout = (const float*)d_in[14];
  const float* bout = (const float*)d_in[15];

  float* logits = (float*)d_out;
  float* ents   = logits + (size_t)TOKENS * VOCAB;

  char* p = (char*)d_ws;
  float* h   = (float*)p;  p += (size_t)TOKENS * D_MODEL * 4;
  u16*   hln = (u16*)p;    p += (size_t)TOKENS * D_MODEL * 2;
  float* qf  = (float*)p;  p += (size_t)TOKENS * D_MODEL * 4;
  float* kf  = (float*)p;  p += (size_t)TOKENS * D_MODEL * 4;
  float* vf  = (float*)p;  p += (size_t)TOKENS * D_MODEL * 4;
  u16*   qbu = (u16*)p;    p += (size_t)TOKENS * D_MODEL * 2;
  u16*   kbu = (u16*)p;    p += (size_t)TOKENS * D_MODEL * 2;
  u16*   vbu = (u16*)p;    p += (size_t)TOKENS * D_MODEL * 2;
  u16*   obu = (u16*)p;    p += (size_t)TOKENS * D_MODEL * 2;
  u16*   mid = (u16*)p;    p += (size_t)TOKENS * D_FF * 2;
  // bf16 weight caches (TDM path)
  const size_t PROJ_W = (size_t)6 * D_MODEL * D_MODEL;   // per Wq/Wk/Wv/Wo
  const size_t FF_W   = (size_t)6 * D_FF * D_MODEL;      // per W1/W2
  const size_t OUT_W  = (size_t)VOCAB * D_MODEL;
  u16* wqb = (u16*)p;  p += PROJ_W * 2;
  u16* wkb = (u16*)p;  p += PROJ_W * 2;
  u16* wvb = (u16*)p;  p += PROJ_W * 2;
  u16* wob = (u16*)p;  p += PROJ_W * 2;
  u16* w1b = (u16*)p;  p += FF_W * 2;
  u16* w2b = (u16*)p;  p += FF_W * 2;
  u16* wub = (u16*)p;  p += OUT_W * 2;
  const size_t NEEDED = (size_t)(p - (char*)d_ws);
  const bool use_tdm = (ws_size >= NEEDED);

  embed_kernel<<<TOKENS, 256, 0, stream>>>(x, emb, h);

  if (use_tdm) {
    cast_kernel<<<PROJ_W / 1024, 256, 0, stream>>>(Wq, wqb);
    cast_kernel<<<PROJ_W / 1024, 256, 0, stream>>>(Wk, wkb);
    cast_kernel<<<PROJ_W / 1024, 256, 0, stream>>>(Wv, wvb);
    cast_kernel<<<PROJ_W / 1024, 256, 0, stream>>>(Wo, wob);
    cast_kernel<<<FF_W   / 1024, 256, 0, stream>>>(W1, w1b);
    cast_kernel<<<FF_W   / 1024, 256, 0, stream>>>(W2, w2b);
    cast_kernel<<<OUT_W  / 1024, 256, 0, stream>>>(Wout, wub);
  }

  const int order[12] = {0, 1, 2, 0, 1, 2, 0, 1, 2, 3, 4, 5};
  const dim3 gProjT(TOKENS / 128, D_MODEL / 128);   // 16 x 8
  const dim3 gFFT(TOKENS / 128, D_FF / 128);        // 16 x 32
  const dim3 gOutT(TOKENS / 128, VOCAB / 128);      // 16 x 250
  const dim3 gProj(TOKENS / BM, D_MODEL / BN);
  const dim3 gFF(TOKENS / BM, D_FF / BN);
  const dim3 gOut(TOKENS / BM, VOCAB / BN);

  for (int e = 0; e < 12; ++e) {
    int i = order[e];
    size_t po = (size_t)i * D_MODEL * D_MODEL;
    size_t fo = (size_t)i * D_FF * D_MODEL;

    ln_kernel<<<TOKENS, 256, 0, stream>>>(h, ln1w + i * D_MODEL, ln1b + i * D_MODEL, hln);
    if (use_tdm) {
      gemm_tdm_kernel<false,false,false,false><<<gProjT, 256, TDM_SMEM_BYTES, stream>>>(
          hln, wqb + po, nullptr, nullptr, qf, TOKENS, D_MODEL, D_MODEL);
      gemm_tdm_kernel<false,false,false,false><<<gProjT, 256, TDM_SMEM_BYTES, stream>>>(
          hln, wkb + po, nullptr, nullptr, kf, TOKENS, D_MODEL, D_MODEL);
      gemm_tdm_kernel<false,false,false,false><<<gProjT, 256, TDM_SMEM_BYTES, stream>>>(
          hln, wvb + po, nullptr, nullptr, vf, TOKENS, D_MODEL, D_MODEL);
    } else {
      gemm_bf16_kernel<false,false,false,false><<<gProj, 128, 0, stream>>>(
          hln, Wq + po, nullptr, nullptr, qf, TOKENS, D_MODEL, D_MODEL);
      gemm_bf16_kernel<false,false,false,false><<<gProj, 128, 0, stream>>>(
          hln, Wk + po, nullptr, nullptr, kf, TOKENS, D_MODEL, D_MODEL);
      gemm_bf16_kernel<false,false,false,false><<<gProj, 128, 0, stream>>>(
          hln, Wv + po, nullptr, nullptr, vf, TOKENS, D_MODEL, D_MODEL);
    }
    rope_kernel<<<(TOKENS * N_HEADS * 32) / 256, 256, 0, stream>>>(
        qf, kf, vf, qbu, kbu, vbu);
    attn_kernel<<<BATCH * N_HEADS, 256, ATTN_SMEM_BYTES, stream>>>(
        qbu, kbu, vbu, obu, ents + (size_t)e * TOKENS * N_HEADS);
    if (use_tdm) {
      gemm_tdm_kernel<false,false,true,false><<<gProjT, 256, TDM_SMEM_BYTES, stream>>>(
          obu, wob + po, nullptr, h, h, TOKENS, D_MODEL, D_MODEL);
    } else {
      gemm_bf16_kernel<false,false,true,false><<<gProj, 128, 0, stream>>>(
          obu, Wo + po, nullptr, h, h, TOKENS, D_MODEL, D_MODEL);
    }
    ln_kernel<<<TOKENS, 256, 0, stream>>>(h, ln2w + i * D_MODEL, ln2b + i * D_MODEL, hln);
    if (use_tdm) {
      gemm_tdm_kernel<true,true,false,true><<<gFFT, 256, TDM_SMEM_BYTES, stream>>>(
          hln, w1b + fo, b1 + (size_t)i * D_FF, nullptr, mid, TOKENS, D_FF, D_MODEL);
      gemm_tdm_kernel<true,false,true,false><<<gProjT, 256, TDM_SMEM_BYTES, stream>>>(
          mid, w2b + fo, b2 + (size_t)i * D_MODEL, h, h, TOKENS, D_MODEL, D_FF);
    } else {
      gemm_bf16_kernel<true,true,false,true><<<gFF, 128, 0, stream>>>(
          hln, W1 + fo, b1 + (size_t)i * D_FF, nullptr, mid, TOKENS, D_FF, D_MODEL);
      gemm_bf16_kernel<true,false,true,false><<<gProj, 128, 0, stream>>>(
          mid, W2 + fo, b2 + (size_t)i * D_MODEL, h, h, TOKENS, D_MODEL, D_FF);
    }
  }

  cast_kernel<<<(TOKENS * D_MODEL) / 1024, 256, 0, stream>>>(h, hln);
  if (use_tdm) {
    gemm_tdm_kernel<true,false,false,false><<<gOutT, 256, TDM_SMEM_BYTES, stream>>>(
        hln, wub, bout, nullptr, logits, TOKENS, VOCAB, D_MODEL);
  } else {
    gemm_bf16_kernel<true,false,false,false><<<gOut, 128, 0, stream>>>(
        hln, Wout, bout, nullptr, logits, TOKENS, VOCAB, D_MODEL);
  }
}
